// SelfAttention1D_46411416600619
// MI455X (gfx1250) — matrix operand order repaired and checked by reading, not compiled, not run on hardware
//
#include <hip/hip_runtime.h>
#include <hip/hip_bf16.h>

typedef __attribute__((ext_vector_type(16))) _Float16 v16h;
typedef __attribute__((ext_vector_type(8)))  _Float16 v8h;
typedef __attribute__((ext_vector_type(8)))  float    v8f;
typedef __attribute__((ext_vector_type(4)))  int      v4i;

// ---------------------------------------------------------------------------
// Async global->LDS copy (CDNA5 GLOBAL_LOAD_ASYNC_TO_LDS_B128, ASYNCcnt).
// Builtin signature (from compiler diagnostic): (v4i as1*, v4i as3*, imm, imm).
// Guarded: falls back to load+ds_store if the builtin is absent.
// ---------------------------------------------------------------------------
#if __has_builtin(__builtin_amdgcn_global_load_async_to_lds_b128)
#define ATHENA_USE_ASYNC 1
#else
#define ATHENA_USE_ASYNC 0
#endif

__device__ __forceinline__ void async_copy16(_Float16* lds, const _Float16* g) {
#if ATHENA_USE_ASYNC
  __builtin_amdgcn_global_load_async_to_lds_b128(
      (__attribute__((address_space(1))) v4i*)(uintptr_t)g,
      (__attribute__((address_space(3))) v4i*)(unsigned int)(uintptr_t)lds,
      0, 0);
#else
  *(uint4*)lds = *(const uint4*)g;
#endif
}

__device__ __forceinline__ void wait_async_all() {
#if ATHENA_USE_ASYNC
#if __has_builtin(__builtin_amdgcn_s_wait_asynccnt)
  __builtin_amdgcn_s_wait_asynccnt(0);
#else
  asm volatile("s_wait_asynccnt 0x0" ::: "memory");
#endif
#endif
}

// ---------------------------------------------------------------------------
// WMMA fragment loaders (CDNA5 ISA 7.12.2 VGPR striping, wave32)
// A (16x32 f16): lanes 0-15 -> M=lane, K 0-7 & 16-23 ; lanes 16-31 -> K 8-15 & 24-31
// B (32x16 f16): lane (mod 16) = N column; lanes 0-15 hold K 0-7 & 16-23, lanes 16-31 K 8-15 & 24-31
// ---------------------------------------------------------------------------
__device__ __forceinline__ v16h load_frag_A(const _Float16* base, int ldk, int lane) {
  int row = lane & 15;
  int kc  = (lane >> 4) * 8;
  v8h a0 = *(const v8h*)(base + row * ldk + kc);
  v8h a1 = *(const v8h*)(base + row * ldk + kc + 16);
  v16h r;
  #pragma unroll
  for (int i = 0; i < 8; i++) { r[i] = a0[i]; r[i + 8] = a1[i]; }
  return r;
}

__device__ __forceinline__ v16h load_frag_B(const _Float16* base, int ldk, int lane) {
  int n  = lane & 15;
  int kc = (lane >> 4) * 8;
  v8h b0 = *(const v8h*)(base + n * ldk + kc);
  v8h b1 = *(const v8h*)(base + n * ldk + kc + 16);
  v16h r;
  #pragma unroll
  for (int i = 0; i < 8; i++) { r[i] = b0[i]; r[i + 8] = b1[i]; }
  return r;
}

// ---------------------------------------------------------------------------
// fp32 -> f16 conversion (weights, once per launch)
// ---------------------------------------------------------------------------
__global__ void cvt_f16_kernel(const float* __restrict__ src, _Float16* __restrict__ dst, int n) {
  int i = (blockIdx.x * blockDim.x + threadIdx.x) * 4;
  if (i < n) {
    float4 v = *(const float4*)(src + i);
    dst[i + 0] = (_Float16)v.x; dst[i + 1] = (_Float16)v.y;
    dst[i + 2] = (_Float16)v.z; dst[i + 3] = (_Float16)v.w;
  }
}

// ---------------------------------------------------------------------------
// LayerNorm (fp32 math) -> f16 output. One row (D=1024) per 256-thread block.
// ---------------------------------------------------------------------------
__global__ __launch_bounds__(256) void ln_f16_kernel(
    const float* __restrict__ x, const float* __restrict__ w, const float* __restrict__ b,
    _Float16* __restrict__ out, int D) {
  __shared__ float red[16];
  int row = blockIdx.x;
  int t = threadIdx.x;
  const float* xr = x + (size_t)row * D;
  float4 v = ((const float4*)xr)[t];
  float s  = v.x + v.y + v.z + v.w;
  float ss = v.x * v.x + v.y * v.y + v.z * v.z + v.w * v.w;
  #pragma unroll
  for (int off = 16; off > 0; off >>= 1) {
    s  += __shfl_down(s,  off, 32);
    ss += __shfl_down(ss, off, 32);
  }
  int wv = t >> 5, ln = t & 31;
  if (ln == 0) { red[wv] = s; red[8 + wv] = ss; }
  __syncthreads();
  if (t == 0) {
    float ts = 0.f, tss = 0.f;
    #pragma unroll
    for (int i = 0; i < 8; i++) { ts += red[i]; tss += red[8 + i]; }
    red[0] = ts; red[8] = tss;
  }
  __syncthreads();
  float mean = red[0] / (float)D;
  float var  = red[8] / (float)D - mean * mean;
  float inv  = rsqrtf(var + 1e-5f);
  float4 wv4 = ((const float4*)w)[t];
  float4 bv4 = ((const float4*)b)[t];
  _Float16* o = out + (size_t)row * D + t * 4;
  o[0] = (_Float16)((v.x - mean) * inv * wv4.x + bv4.x);
  o[1] = (_Float16)((v.y - mean) * inv * wv4.y + bv4.y);
  o[2] = (_Float16)((v.z - mean) * inv * wv4.z + bv4.z);
  o[3] = (_Float16)((v.w - mean) * inv * wv4.w + bv4.w);
}

// ---------------------------------------------------------------------------
// Tiled GEMM: out[M,N] = A[M,K](f16) * W[N,K]^T(f16) + bias, fused epilogues.
// 256 threads = 8 waves; block tile 128x128; each wave 32x64 (2x4 WMMA).
// Double-buffered LDS, filled with async global->LDS copies (ASYNCcnt).
// MODE 0: f16 out, scale cols < scale_nlim (fused QK scaling)
// MODE 1: f32 out = resid + y   (residual add)
// MODE 2: f16 out = gelu_exact(y)
// ---------------------------------------------------------------------------
template <int MODE>
__global__ __launch_bounds__(256) void gemm_wmma(
    const _Float16* __restrict__ A, const _Float16* __restrict__ W,
    const float* __restrict__ bias, const float* __restrict__ resid,
    float scale, int scale_nlim, void* __restrict__ outp,
    int M, int N, int K) {
  __shared__ __align__(16) _Float16 As[2][128 * 32];
  __shared__ __align__(16) _Float16 Bs[2][128 * 32];
  int n0 = blockIdx.x * 128, m0 = blockIdx.y * 128;
  int t = threadIdx.x, lane = t & 31, wave = t >> 5;
  int wm = (wave >> 1) * 32;   // 4 wave rows   -> 128 M
  int wn = (wave & 1) * 64;    // 2 wave cols   -> 128 N
  int lrow = t >> 1, lc = (t & 1) * 16;

  const _Float16* gA = A + (size_t)(m0 + lrow) * K + lc;
  const _Float16* gW = W + (size_t)(n0 + lrow) * K + lc;
  _Float16* lA = &As[0][0] + lrow * 32 + lc;
  _Float16* lB = &Bs[0][0] + lrow * 32 + lc;
  const int bufStride = 128 * 32;

  auto issue = [&](int k, int buf) {
    const _Float16* sa = gA + k * 32;
    const _Float16* sb = gW + k * 32;
    async_copy16(lA + buf * bufStride,     sa);
    async_copy16(lA + buf * bufStride + 8, sa + 8);
    async_copy16(lB + buf * bufStride,     sb);
    async_copy16(lB + buf * bufStride + 8, sb + 8);
  };

  v8f acc[2][4] = {};
  int nk = K / 32;
  issue(0, 0);

  for (int k = 0; k < nk; ++k) {
    int cur = k & 1;
    wait_async_all();      // stage-k tiles resident (this wave's copies)
    __syncthreads();       // ... and everyone else's
    if (k + 1 < nk) issue(k + 1, cur ^ 1);   // prefetch overlaps the math

    const _Float16* as = &As[cur][0];
    const _Float16* bs = &Bs[cur][0];
    v16h af[2], bf[4];
    #pragma unroll
    for (int mi = 0; mi < 2; mi++) af[mi] = load_frag_A(as + (wm + mi * 16) * 32, 32, lane);
    #pragma unroll
    for (int ni = 0; ni < 4; ni++) bf[ni] = load_frag_B(bs + (wn + ni * 16) * 32, 32, lane);
    #pragma unroll
    for (int mi = 0; mi < 2; mi++)
      #pragma unroll
      for (int ni = 0; ni < 4; ni++)
        acc[mi][ni] = __builtin_amdgcn_wmma_f32_16x16x32_f16(
            false, af[mi], false, bf[ni], (short)0, acc[mi][ni], false, false);
    __syncthreads();       // all waves done with buffer `cur` before it is refilled
  }

  int coff = lane & 15;
  int madd = (lane & 16) ? 8 : 0;
  #pragma unroll
  for (int mi = 0; mi < 2; mi++)
    #pragma unroll
    for (int ni = 0; ni < 4; ni++) {
      int colg = n0 + wn + ni * 16 + coff;
      float bv = bias[colg];
      #pragma unroll
      for (int r = 0; r < 8; r++) {
        int rowg = m0 + wm + mi * 16 + madd + r;
        float v = acc[mi][ni][r] + bv;
        size_t idx = (size_t)rowg * N + colg;
        if (MODE == 0) {
          if (colg < scale_nlim) v *= scale;
          ((_Float16*)outp)[idx] = (_Float16)v;
        } else if (MODE == 1) {
          ((float*)outp)[idx] = resid[idx] + v;
        } else {
          float g = 0.5f * v * (1.0f + erff(v * 0.70710678118f));
          ((_Float16*)outp)[idx] = (_Float16)g;
        }
      }
    }
}

// ---------------------------------------------------------------------------
// Flash attention: one (b,h) x 64-query tile per 128-thread block.
// Q,K pre-scaled by 64^-1/4 in the QKV GEMM epilogue.
// qkv layout: [B, L, 3D] f16. out: [B*L, D] f16 (feeds proj GEMM).
// ---------------------------------------------------------------------------
__global__ __launch_bounds__(128) void attn_wmma(
    const _Float16* __restrict__ qkv, const int* __restrict__ mask,
    _Float16* __restrict__ out, int Bn, int Ln, int Dn, int Hn) {
  const int CH = 64;
  __shared__ __align__(16) _Float16 Qs[64 * 64];
  __shared__ __align__(16) _Float16 Ks[64 * 64];
  __shared__ __align__(16) _Float16 Vts[64 * 64];   // V transposed [ch][key]
  __shared__ __align__(16) float    Ss[64 * 64];
  __shared__ __align__(16) _Float16 Ps[64 * 64];
  __shared__ float mrow[64], lsum[64], crow[64];
  __shared__ int smask[64];

  int qt = blockIdx.x, bh = blockIdx.y;
  int b = bh / Hn, h = bh % Hn;
  int t = threadIdx.x, lane = t & 31, wave = t >> 5;
  int q0 = qt * 64;
  size_t tok = (size_t)3 * Dn;
  const _Float16* qbase = qkv + (size_t)b * Ln * tok + h * CH;
  const _Float16* kbase = qbase + Dn;
  const _Float16* vbase = qbase + 2 * Dn;

  {  // load Q tile [64 x 64] straight to LDS (async path when available)
    int r = t >> 1, c0 = (t & 1) * 32;
    const _Float16* src = qbase + (size_t)(q0 + r) * tok + c0;
    _Float16* dst = Qs + r * 64 + c0;
    async_copy16(dst, src);
    async_copy16(dst + 8,  src + 8);
    async_copy16(dst + 16, src + 16);
    async_copy16(dst + 24, src + 24);
  }
  if (t < 64) { mrow[t] = -3.0e38f; lsum[t] = 0.0f; }

  int mtile = wave * 16;
  int coff = lane & 15;
  int madd = (lane & 16) ? 8 : 0;
  v8f o[4] = {};

  for (int s0 = 0; s0 < Ln; s0 += 64) {
    wait_async_all();
    __syncthreads();  // previous iteration done with Ks/Vts/Ps (and Q resident)
    {  // load K tile (async) + V tile transposed through VGPRs
      int r = t >> 1, c0 = (t & 1) * 32;
      const _Float16* ksrc = kbase + (size_t)(s0 + r) * tok + c0;
      _Float16* kdst = Ks + r * 64 + c0;
      async_copy16(kdst,      ksrc);
      async_copy16(kdst + 8,  ksrc + 8);
      async_copy16(kdst + 16, ksrc + 16);
      async_copy16(kdst + 24, ksrc + 24);
      uint4 vv[4];
      const uint4* vsrc = (const uint4*)(vbase + (size_t)(s0 + r) * tok + c0);
      vv[0] = vsrc[0]; vv[1] = vsrc[1]; vv[2] = vsrc[2]; vv[3] = vsrc[3];
      const _Float16* ve = (const _Float16*)vv;
      #pragma unroll
      for (int j = 0; j < 32; j++) Vts[(c0 + j) * 64 + r] = ve[j];
    }
    if (t < 64) smask[t] = mask[b * Ln + s0 + t];
    wait_async_all();
    __syncthreads();

    // S = Q K^T : 16 query rows x 64 keys per wave (2 k-steps x 4 key tiles)
    v8f sacc[4] = {};
    #pragma unroll
    for (int ks = 0; ks < 2; ks++) {
      v16h aq = load_frag_A(Qs + mtile * 64 + ks * 32, 64, lane);
      #pragma unroll
      for (int st = 0; st < 4; st++) {
        v16h bk = load_frag_B(Ks + (st * 16) * 64 + ks * 32, 64, lane);
        sacc[st] = __builtin_amdgcn_wmma_f32_16x16x32_f16(
            false, aq, false, bk, (short)0, sacc[st], false, false);
      }
    }
    #pragma unroll
    for (int st = 0; st < 4; st++)
      #pragma unroll
      for (int r = 0; r < 8; r++)
        Ss[(mtile + madd + r) * 64 + st * 16 + coff] = sacc[st][r];
    __syncthreads();

    // online softmax update, one row per thread (threads 0..63)
    if (t < 64) {
      float mold = mrow[t];
      float mx = mold;
      for (int j = 0; j < 64; j++) {
        float sv = (smask[j] == 0) ? -3.0e38f : Ss[t * 64 + j];
        mx = fmaxf(mx, sv);
      }
      float corr = __expf(mold - mx);
      float sum = 0.0f;
      for (int j = 0; j < 64; j++) {
        float sv = (smask[j] == 0) ? -3.0e38f : Ss[t * 64 + j];
        float p = __expf(sv - mx);
        Ps[t * 64 + j] = (_Float16)p;
        sum += p;
      }
      mrow[t] = mx;
      lsum[t] = lsum[t] * corr + sum;
      crow[t] = corr;
    }
    __syncthreads();

    // rescale running output, then O += P V
    float cr[8];
    #pragma unroll
    for (int r = 0; r < 8; r++) cr[r] = crow[mtile + madd + r];
    #pragma unroll
    for (int nt = 0; nt < 4; nt++)
      #pragma unroll
      for (int r = 0; r < 8; r++) o[nt][r] *= cr[r];
    #pragma unroll
    for (int ks = 0; ks < 2; ks++) {
      v16h ap = load_frag_A(Ps + mtile * 64 + ks * 32, 64, lane);
      #pragma unroll
      for (int nt = 0; nt < 4; nt++) {
        v16h bv = load_frag_B(Vts + (nt * 16) * 64 + ks * 32, 64, lane);
        o[nt] = __builtin_amdgcn_wmma_f32_16x16x32_f16(
            false, ap, false, bv, (short)0, o[nt], false, false);
      }
    }
  }
  __syncthreads();

  float linv[8];
  #pragma unroll
  for (int r = 0; r < 8; r++) linv[r] = 1.0f / lsum[mtile + madd + r];
  #pragma unroll
  for (int nt = 0; nt < 4; nt++)
    #pragma unroll
    for (int r = 0; r < 8; r++) {
      int rowg = q0 + mtile + madd + r;
      int colg = h * CH + nt * 16 + coff;
      out[(size_t)(b * Ln + rowg) * Dn + colg] = (_Float16)(o[nt][r] * linv[r]);
    }
}

// ---------------------------------------------------------------------------
extern "C" void kernel_launch(void* const* d_in, const int* in_sizes, int n_in,
                              void* d_out, int out_size, void* d_ws, size_t ws_size,
                              hipStream_t stream) {
  const int Bn = 2, Ln = 2048, Dn = 1024, Hn = 16;
  const int M = Bn * Ln;                       // 4096 tokens
  const float qk_scale = 0.35355339059327373f; // 64^-1/4

  const float* x      = (const float*)d_in[0];
  const int*   mask   = (const int*)d_in[1];
  const float* ln_w   = (const float*)d_in[2];
  const float* ln_b   = (const float*)d_in[3];
  const float* qkv_w  = (const float*)d_in[4];
  const float* qkv_b  = (const float*)d_in[5];
  const float* proj_w = (const float*)d_in[6];
  const float* proj_b = (const float*)d_in[7];
  const float* ln2_w  = (const float*)d_in[8];
  const float* ln2_b  = (const float*)d_in[9];
  const float* w1     = (const float*)d_in[10];
  const float* b1     = (const float*)d_in[11];
  const float* w2     = (const float*)d_in[12];
  const float* b2     = (const float*)d_in[13];

  char* wp = (char*)d_ws;
  auto take = [&](size_t nbytes) {
    void* p = (void*)wp;
    wp += (nbytes + 255) & ~(size_t)255;
    return p;
  };
  _Float16* xh   = (_Float16*)take((size_t)M * Dn * 2);
  _Float16* wqh  = (_Float16*)take((size_t)3 * Dn * Dn * 2);
  _Float16* wph  = (_Float16*)take((size_t)Dn * Dn * 2);
  _Float16* w1h  = (_Float16*)take((size_t)4 * Dn * Dn * 2);
  _Float16* w2h  = (_Float16*)take((size_t)4 * Dn * Dn * 2);
  _Float16* qkvh = (_Float16*)take((size_t)M * 3 * Dn * 2);
  _Float16* ah   = (_Float16*)take((size_t)M * Dn * 2);
  float*    x2   = (float*)take((size_t)M * Dn * 4);
  _Float16* h16  = (_Float16*)take((size_t)M * Dn * 2);
  _Float16* h1h  = (_Float16*)take((size_t)M * 4 * Dn * 2);

  // 1) downconvert weights to f16
  auto cvt = [&](const float* s, _Float16* d, int n) {
    cvt_f16_kernel<<<(n / 4 + 255) / 256, 256, 0, stream>>>(s, d, n);
  };
  cvt(qkv_w, wqh, 3 * Dn * Dn);
  cvt(proj_w, wph, Dn * Dn);
  cvt(w1, w1h, 4 * Dn * Dn);
  cvt(w2, w2h, 4 * Dn * Dn);

  // 2) LN1 -> f16
  ln_f16_kernel<<<M, 256, 0, stream>>>(x, ln_w, ln_b, xh, Dn);

  // 3) QKV = LN(x) @ qkv_w^T + b, fold 64^-1/4 into Q and K columns (< 2D)
  gemm_wmma<0><<<dim3(3 * Dn / 128, M / 128), 256, 0, stream>>>(
      xh, wqh, qkv_b, nullptr, qk_scale, 2 * Dn, (void*)qkvh, M, 3 * Dn, Dn);

  // 4) flash attention
  attn_wmma<<<dim3(Ln / 64, Bn * Hn), 128, 0, stream>>>(qkvh, mask, ah, Bn, Ln, Dn, Hn);

  // 5) x2 = x + attn @ proj_w^T + proj_b  (fp32)
  gemm_wmma<1><<<dim3(Dn / 128, M / 128), 256, 0, stream>>>(
      ah, wph, proj_b, x, 1.0f, 0, (void*)x2, M, Dn, Dn);

  // 6) LN2 -> f16
  ln_f16_kernel<<<M, 256, 0, stream>>>(x2, ln2_w, ln2_b, h16, Dn);

  // 7) h1 = gelu(LN2 @ w1^T + b1)  (f16)
  gemm_wmma<2><<<dim3(4 * Dn / 128, M / 128), 256, 0, stream>>>(
      h16, w1h, b1, nullptr, 1.0f, 0, (void*)h1h, M, 4 * Dn, Dn);

  // 8) out = x2 + h1 @ w2^T + b2  (fp32)
  gemm_wmma<1><<<dim3(Dn / 128, M / 128), 256, 0, stream>>>(
      h1h, w2h, b2, x2, 1.0f, 0, d_out, M, Dn, 4 * Dn);
}
